// MambaAdapter_64201171140745
// MI455X (gfx1250) — matrix operand order repaired and checked by reading, not compiled, not run on hardware
//
#include <hip/hip_runtime.h>

typedef __attribute__((ext_vector_type(16))) __bf16 bf16x16;
typedef __attribute__((ext_vector_type(8)))  __bf16 bf16x8;
typedef __attribute__((ext_vector_type(8)))  float  f32x8;
typedef unsigned short u16;

#define DIM     768
#define D_INNER 768
#define KK      16
#define CD      48
#define NS      16
#define BATCH   8
#define SEQ     4096
#define MROWS   (BATCH * SEQ)   // 32768

__device__ __forceinline__ u16 f32_to_bf16(float f) {
  unsigned int u = __float_as_uint(f);
  u += 0x7FFFu + ((u >> 16) & 1u);   // round-to-nearest-even
  return (u16)(u >> 16);
}

// ---------------------------------------------------------------- convert
__global__ __launch_bounds__(256)
void cvt_f32_to_bf16_kernel(const float* __restrict__ in, u16* __restrict__ out, int n) {
  int i = blockIdx.x * 256 + threadIdx.x;
  if (i < n) out[i] = f32_to_bf16(in[i]);
}

// ---------------------------------------------------------------- async copy helpers
// CDNA5 async global->LDS (ASYNCcnt-tracked, no VGPR round trip).
// dest is the LDS byte address (generic shared ptr truncated to addr[31:0]).
__device__ __forceinline__ void async_ld_b128(unsigned lds_off, const void* gaddr) {
  asm volatile("global_load_async_to_lds_b128 %0, %1, off"
               :: "v"(lds_off), "v"(gaddr) : "memory");
}
__device__ __forceinline__ unsigned lds_addr(const void* p) {
  return (unsigned)(unsigned long long)p;   // low 32 bits = LDS byte address
}

// ---------------------------------------------------------------- bf16 WMMA GEMM
// C[m,n] = sum_k A[m,k] * B[n,k] + bias[n]
// A: M x K row-major bf16, B: N x K row-major bf16 (i.e. W), C: M x N f32.
#define BM 128
#define BN 64
#define BK 32
#define LDA 40   // padded LDS row (bf16 elems) -> 80B, conflict-free b128 sweeps
#define LDB 40

union FragU { bf16x16 v; bf16x8 h[2]; };

__global__ __launch_bounds__(256)
void gemm_bf16_wmma_kernel(const u16* __restrict__ A, const u16* __restrict__ B,
                           const float* __restrict__ bias, float* __restrict__ C,
                           int M, int N, int K)
{
  __shared__ u16 As[2][BM * LDA];
  __shared__ u16 Bs[2][BN * LDB];

  const int tid  = threadIdx.x;
  const int lane = tid & 31;
  const int wave = tid >> 5;
  const int wm   = wave >> 1;     // 0..3  (M strip of 32)
  const int wn   = wave & 1;      // 0..1  (N strip of 32)
  const int bm   = blockIdx.x * BM;
  const int bn   = blockIdx.y * BN;

  const int lm = lane & 15;
  const int hi = lane >> 4;

  f32x8 acc[2][2] = {};

  // global->LDS mapping: A 128x32 (2 threads/row, 2x16B), B 64x32 (4 threads/row, 1x16B)
  const int ar = tid >> 1;
  const int ac = (tid & 1) << 4;
  const int br = tid >> 2;
  const int bc = (tid & 3) << 3;

  const u16* gA = A + (size_t)(bm + ar) * K + ac;
  const u16* gB = B + (size_t)(bn + br) * K + bc;

  unsigned ldsA[2], ldsB[2];
  #pragma unroll
  for (int s = 0; s < 2; ++s) {
    ldsA[s] = lds_addr(&As[s][ar * LDA + ac]);
    ldsB[s] = lds_addr(&Bs[s][br * LDB + bc]);
  }

  // prologue: tile 0 -> buffer 0 (3 async b128 copies per thread)
  async_ld_b128(ldsA[0],      gA);
  async_ld_b128(ldsA[0] + 16, gA + 8);
  async_ld_b128(ldsB[0],      gB);

  const int NK = K / BK;
  for (int it = 0; it < NK; ++it) {
    const int cur = it & 1;
    if (it + 1 < NK) {
      const u16* nA = gA + (it + 1) * BK;
      const u16* nB = gB + (it + 1) * BK;
      async_ld_b128(ldsA[cur ^ 1],      nA);      // tile it+1 in flight
      async_ld_b128(ldsA[cur ^ 1] + 16, nA + 8);  // while tile it computes
      async_ld_b128(ldsB[cur ^ 1],      nB);
      asm volatile("s_wait_asynccnt 0x3" ::: "memory");  // tile `it` landed
    } else {
      asm volatile("s_wait_asynccnt 0x0" ::: "memory");
    }
    __syncthreads();               // tile `it` resident for all waves

    // A fragment: lane(m=lm, hi): elems[0..7]=K[hi*8..+8), elems[8..15]=K[16+hi*8..+8)
    FragU af[2], bfm[2];
    #pragma unroll
    for (int i = 0; i < 2; ++i) {
      int row = wm * 32 + i * 16 + lm;
      af[i].h[0] = *(bf16x8*)&As[cur][row * LDA + hi * 8];
      af[i].h[1] = *(bf16x8*)&As[cur][row * LDA + 16 + hi * 8];
    }
    // B fragment: lane(n=lm, hi): elems[0..7]=K[hi*8..+8), elems[8..15]=K[16+hi*8..+8)
    #pragma unroll
    for (int j = 0; j < 2; ++j) {
      int col = wn * 32 + j * 16 + lm;
      bfm[j].h[0] = *(bf16x8*)&Bs[cur][col * LDB + hi * 8];
      bfm[j].h[1] = *(bf16x8*)&Bs[cur][col * LDB + 16 + hi * 8];
    }
    #pragma unroll
    for (int i = 0; i < 2; ++i)
      #pragma unroll
      for (int j = 0; j < 2; ++j)
        acc[i][j] = __builtin_amdgcn_wmma_f32_16x16x32_bf16(
            false, af[i].v, false, bfm[j].v, (short)0, acc[i][j], false, false);

    __syncthreads();               // all reads done before buffer reuse
  }

  // D layout: VGPR v holds (M = v + 8*hi, N = lm)
  #pragma unroll
  for (int i = 0; i < 2; ++i) {
    #pragma unroll
    for (int j = 0; j < 2; ++j) {
      int col   = bn + wn * 32 + j * 16 + lm;
      float bv  = bias[col];
      int rbase = bm + wm * 32 + i * 16 + hi * 8;
      #pragma unroll
      for (int v = 0; v < 8; ++v)
        C[(size_t)(rbase + v) * N + col] = acc[i][j][v] + bv;
    }
  }
}

// ---------------------------------------------------------------- fused middle
// y[b,c,l] = softplus(delta+db)^2 * silu(conv3(u)) * sum_n A[k,c',n]*B_t*C_t
// Writes Y in (b, l, c) layout as bf16, ready as GEMM2's A operand.
#define LT 64
__global__ __launch_bounds__(256)
void mamba_middle_kernel(const float* __restrict__ U, const float* __restrict__ delta,
                         const float* __restrict__ Bt, const float* __restrict__ Ct,
                         const float* __restrict__ A_logs, const float* __restrict__ conv_w,
                         const float* __restrict__ conv_b, const float* __restrict__ delta_bias,
                         u16* __restrict__ Y)
{
  __shared__ float BC[NS][LT];
  __shared__ float Ash[CD * NS];
  const int bk = blockIdx.x;          // b*16 + k
  const int b  = bk >> 4;
  const int k  = bk & 15;
  const int l0 = blockIdx.y * LT;
  const int t  = threadIdx.x;

  const size_t bcbase = (size_t)bk * NS * SEQ;     // B_t[b][k][.][.]
  for (int idx = t; idx < NS * LT; idx += 256) {
    int n = idx / LT, l = idx % LT;
    size_t g = bcbase + (size_t)n * SEQ + l0 + l;
    BC[n][l] = Bt[g] * Ct[g];
  }
  for (int idx = t; idx < CD * NS; idx += 256)
    Ash[idx] = A_logs[k * CD * NS + idx];
  __syncthreads();

  for (int idx = t; idx < CD * LT; idx += 256) {
    int cp = idx % CD;                 // c' fastest -> U accesses coalesce
    int l  = idx / CD;
    int c  = k * CD + cp;
    int lg = l0 + l;

    float s = 0.f;
    #pragma unroll
    for (int n = 0; n < NS; ++n) s += Ash[cp * NS + n] * BC[n][l];

    // depthwise 3-tap conv along l (only middle column of 3x3 survives)
    size_t urow = ((size_t)b * SEQ + lg) * D_INNER + c;
    float um = (lg > 0)       ? U[urow - D_INNER] : 0.f;
    float u0 = U[urow];
    float up = (lg < SEQ - 1) ? U[urow + D_INNER] : 0.f;
    const float* w = conv_w + c * 9;   // w[c,0,kh,1] at c*9 + kh*3 + 1
    float uc = w[1] * um + w[4] * u0 + w[7] * up + conv_b[c];
    float sg = 1.f / (1.f + __expf(-uc));
    float usil = uc * sg;

    float dv = delta[((size_t)b * (KK * CD) + c) * SEQ + lg] + delta_bias[c];
    float d  = (dv > 20.f) ? dv : log1pf(__expf(dv));

    float y = d * d * usil * s;
    Y[urow] = f32_to_bf16(y);
  }
}

// ---------------------------------------------------------------- launch
extern "C" void kernel_launch(void* const* d_in, const int* in_sizes, int n_in,
                              void* d_out, int out_size, void* d_ws, size_t ws_size,
                              hipStream_t stream)
{
  const float* x          = (const float*)d_in[0];
  const float* in_proj_w  = (const float*)d_in[1];
  const float* in_proj_b  = (const float*)d_in[2];
  const float* conv_w     = (const float*)d_in[3];
  const float* conv_b     = (const float*)d_in[4];
  const float* out_proj_w = (const float*)d_in[5];
  const float* out_proj_b = (const float*)d_in[6];
  const float* A_logs     = (const float*)d_in[7];
  const float* delta      = (const float*)d_in[8];
  const float* B_t        = (const float*)d_in[9];
  const float* C_t        = (const float*)d_in[10];
  const float* delta_bias = (const float*)d_in[11];
  float* out = (float*)d_out;

  // workspace carve-up (~204 MB)
  char* p = (char*)d_ws;
  u16*   Xbf = (u16*)p;   p += (size_t)MROWS * DIM     * sizeof(u16);
  u16*   W1  = (u16*)p;   p += (size_t)D_INNER * DIM   * sizeof(u16);
  u16*   W2  = (u16*)p;   p += (size_t)DIM * D_INNER   * sizeof(u16);
  float* U   = (float*)p; p += (size_t)MROWS * D_INNER * sizeof(float);
  u16*   Ybf = (u16*)p;

  int nx = MROWS * DIM;
  cvt_f32_to_bf16_kernel<<<(nx + 255) / 256, 256, 0, stream>>>(x, Xbf, nx);
  int nw = D_INNER * DIM;   // only first 768 rows of in_proj_w are ever used
  cvt_f32_to_bf16_kernel<<<(nw + 255) / 256, 256, 0, stream>>>(in_proj_w, W1, nw);
  cvt_f32_to_bf16_kernel<<<(nw + 255) / 256, 256, 0, stream>>>(out_proj_w, W2, nw);

  dim3 g1(MROWS / BM, D_INNER / BN);
  gemm_bf16_wmma_kernel<<<g1, 256, 0, stream>>>(Xbf, W1, in_proj_b, U,
                                                MROWS, D_INNER, DIM);

  dim3 gm(BATCH * KK, SEQ / LT);
  mamba_middle_kernel<<<gm, 256, 0, stream>>>(U, delta, B_t, C_t, A_logs,
                                              conv_w, conv_b, delta_bias, Ybf);

  dim3 g2(MROWS / BM, DIM / BN);
  gemm_bf16_wmma_kernel<<<g2, 256, 0, stream>>>(Ybf, W2, out_proj_b, out,
                                                MROWS, DIM, D_INNER);
}
